// MixtureOfExperts_1623497637920
// MI455X (gfx1250) — matrix operand order repaired and checked by reading, not compiled, not run on hardware
//
#include <hip/hip_runtime.h>
#include <math.h>

// ---------------- problem constants ----------------
constexpr int T_TOK = 8192;
constexpr int NEXP  = 16;
constexpr int DIN   = 1024;
constexpr int DOUT  = 1024;
constexpr int TOPK  = 2;

// GEMM tiling
constexpr int TM = 64;    // rows (gathered token-assignments) per block
constexpr int TN = 128;   // output features per block
constexpr int KT = 64;    // K step staged in LDS
constexpr int ROWS_PAD   = T_TOK * TOPK + NEXP * TM;   // 17408 worst-case padded rows
constexpr int MAX_MTILES = ROWS_PAD / TM;              // 272

// ---------------- small vector types (trivial, union-safe) ----------------
typedef float    f32x4  __attribute__((ext_vector_type(4)));
typedef unsigned u32x4  __attribute__((ext_vector_type(4)));
typedef __bf16   v16bf  __attribute__((ext_vector_type(16)));
typedef float    v8f    __attribute__((ext_vector_type(8)));

union Frag { u32x4 u[2]; v16bf v; };

// fp32 -> bf16 round-to-nearest-even
__device__ __forceinline__ unsigned short f2bf(float f) {
  unsigned u = __builtin_bit_cast(unsigned, f);
  u += 0x7FFFu + ((u >> 16) & 1u);
  return (unsigned short)(u >> 16);
}

// pack 8 fp32 -> 8 bf16 in one 16-byte word
__device__ __forceinline__ u32x4 pack8(f32x4 a, f32x4 b) {
  union { unsigned short s[8]; u32x4 u; } r;
  r.s[0] = f2bf(a.x); r.s[1] = f2bf(a.y); r.s[2] = f2bf(a.z); r.s[3] = f2bf(a.w);
  r.s[4] = f2bf(b.x); r.s[5] = f2bf(b.y); r.s[6] = f2bf(b.z); r.s[7] = f2bf(b.w);
  return r.u;
}

// ---------------- kernel 1: init (zero out, poison row map, zero counts) ----
__global__ void moe_init(float* __restrict__ out, int* __restrict__ row_token,
                         int* __restrict__ counts, long total_out) {
  long i = (long)blockIdx.x * blockDim.x + threadIdx.x;
  if (i < total_out) out[i] = 0.0f;
  if (i < ROWS_PAD)  row_token[i] = -1;
  if (i < NEXP)      counts[i] = 0;
}

// ---------------- kernel 2: router (scores, top-2, softmax, counts) ---------
__global__ void moe_router(const float* __restrict__ x, const float* __restrict__ Wr,
                           const float* __restrict__ br,
                           float* __restrict__ probs,      // [T,2]
                           int*   __restrict__ tok_e,      // [T*2]
                           float* __restrict__ tok_p,      // [T*2]
                           int*   __restrict__ counts) {
  __shared__ float sc[16][NEXP];
  const int tl = threadIdx.x >> 4;        // local token 0..15
  const int e  = threadIdx.x & 15;        // expert 0..15
  const int t  = blockIdx.x * 16 + tl;

  const f32x4* xr = (const f32x4*)(x  + (size_t)t * DIN);
  const f32x4* wr = (const f32x4*)(Wr + (size_t)e * DIN);
  float acc = 0.0f;
  for (int i = 0; i < DIN / 4; ++i) {
    f32x4 a = xr[i], b = wr[i];
    acc += a.x * b.x + a.y * b.y + a.z * b.z + a.w * b.w;
  }
  sc[tl][e] = acc + br[e];
  __syncthreads();

  if (e == 0) {
    // top-2, first occurrence wins ties (matches jax.lax.top_k)
    float b0 = sc[tl][0]; int i0 = 0;
    for (int j = 1; j < NEXP; ++j) { float v = sc[tl][j]; if (v > b0) { b0 = v; i0 = j; } }
    float b1 = -INFINITY; int i1 = 0;
    for (int j = 0; j < NEXP; ++j) {
      if (j == i0) continue;
      float v = sc[tl][j]; if (v > b1) { b1 = v; i1 = j; }
    }
    // softmax over the two selected scores (b0 >= b1)
    float e1 = expf(b1 - b0);
    float inv = 1.0f / (1.0f + e1);
    float p0 = inv, p1 = e1 * inv;
    probs[t * 2 + 0] = p0;  probs[t * 2 + 1] = p1;
    tok_e[t * 2 + 0] = i0;  tok_e[t * 2 + 1] = i1;
    tok_p[t * 2 + 0] = p0;  tok_p[t * 2 + 1] = p1;
    atomicAdd(&counts[i0], 1);
    atomicAdd(&counts[i1], 1);
  }
}

// ---------------- kernel 3: padded prefix-sum offsets -----------------------
__global__ void moe_offsets(const int* __restrict__ counts, int* __restrict__ off,
                            int* __restrict__ cursor) {
  int o = 0;
  for (int e = 0; e < NEXP; ++e) {
    off[e] = o;
    o += (counts[e] + TM - 1) & ~(TM - 1);   // pad each segment to TM rows
    cursor[e] = 0;
  }
  off[NEXP] = o;
}

// ---------------- kernel 4: scatter assignments into expert-grouped rows ----
__global__ void moe_place(const int* __restrict__ tok_e, const float* __restrict__ tok_p,
                          const int* __restrict__ off, int* __restrict__ cursor,
                          int* __restrict__ row_token, float* __restrict__ row_gate) {
  int i = blockIdx.x * blockDim.x + threadIdx.x;    // assignment id, [0, T*2)
  if (i >= T_TOK * TOPK) return;
  int e = tok_e[i];
  int slot = atomicAdd(&cursor[e], 1);
  int row  = off[e] + slot;
  row_token[row] = i >> 1;       // token id
  row_gate[row]  = tok_p[i];     // gate value
}

// ---------------- kernel 5: grouped expert GEMM with bf16 WMMA --------------
__global__ __launch_bounds__(256) void moe_expert_gemm(
    const float* __restrict__ x, const float* __restrict__ We,
    const float* __restrict__ be,
    const int* __restrict__ row_token, const float* __restrict__ row_gate,
    const int* __restrict__ off, float* __restrict__ out) {
  __shared__ unsigned short As[TM][KT];   // gathered token rows (bf16)
  __shared__ unsigned short Bs[TN][KT];   // We rows (== B columns), bf16
  __shared__ int   ltok[TM];
  __shared__ float lgate[TM];

  const int r0 = blockIdx.x * TM;
  const int n0 = blockIdx.y * TN;

  int offs[NEXP + 1];
#pragma unroll
  for (int i = 0; i <= NEXP; ++i) offs[i] = off[i];
  if (r0 >= offs[NEXP]) return;           // uniform early-exit before any barrier
  int e = 0;
#pragma unroll
  for (int i = 0; i < NEXP; ++i) if (offs[i + 1] <= r0) e = i + 1;

  const int tid = threadIdx.x;
  if (tid < TM) { ltok[tid] = row_token[r0 + tid]; lgate[tid] = row_gate[r0 + tid]; }

  const int lane = tid & 31, wave = tid >> 5;
  const int wm = wave >> 2, wn = wave & 3;         // 2x4 wave grid -> 32x32 per wave
  const int sub = lane >> 4, lidx = lane & 15;

  const int arow = tid >> 2, aq = tid & 3;          // A stage: 64 rows x 4 quads of 16
  const int brow = tid >> 1, bh = tid & 1;          // B stage: 128 rows x 2 halves of 32

  v8f acc[2][2] = {};

  const float* wrow = We + ((size_t)e * DOUT + (n0 + brow)) * DIN + bh * 32;

  __syncthreads();
  const int atok = ltok[arow];
  const float* xrow = x + (size_t)(atok < 0 ? 0 : atok) * DIN + aq * 16;

  for (int kb = 0; kb < DIN; kb += KT) {
    // ---- stage A tile (fp32 -> bf16) ----
    u32x4* adst = (u32x4*)&As[arow][aq * 16];
    if (atok >= 0) {
      const f32x4* s = (const f32x4*)(xrow + kb);
      adst[0] = pack8(s[0], s[1]);
      adst[1] = pack8(s[2], s[3]);
    } else {
      u32x4 z = {0u, 0u, 0u, 0u};
      adst[0] = z; adst[1] = z;
    }
    // ---- stage B tile (fp32 -> bf16); We[f,:] is a B column, contiguous in K
    {
      const f32x4* s = (const f32x4*)(wrow + kb);
      u32x4* bdst = (u32x4*)&Bs[brow][bh * 32];
#pragma unroll
      for (int q = 0; q < 4; ++q) bdst[q] = pack8(s[2 * q], s[2 * q + 1]);
    }
    __syncthreads();

    // ---- compute: 2 K-steps of 32, 2x2 WMMA subtiles per wave ----
#pragma unroll
    for (int kk = 0; kk < KT; kk += 32) {
      Frag a[2], b[2];
#pragma unroll
      for (int m2 = 0; m2 < 2; ++m2) {
        // A 16x32 bf16 lane layout: row=lidx, K = kk+8*sub+(0..7) and kk+16+8*sub+(0..7)
        const u32x4* p = (const u32x4*)&As[32 * wm + 16 * m2 + lidx][kk + 8 * sub];
        a[m2].u[0] = p[0];
        a[m2].u[1] = p[2];   // +16 elements
      }
#pragma unroll
      for (int n2 = 0; n2 < 2; ++n2) {
        // B 32x16 bf16 lane layout: col=lidx, K = kk+8*sub+(0..7) and kk+16+8*sub+(0..7)
        const u32x4* p = (const u32x4*)&Bs[32 * wn + 16 * n2 + lidx][kk + 8 * sub];
        b[n2].u[0] = p[0];
        b[n2].u[1] = p[2];
      }
#pragma unroll
      for (int m2 = 0; m2 < 2; ++m2)
#pragma unroll
        for (int n2 = 0; n2 < 2; ++n2)
          acc[m2][n2] = __builtin_amdgcn_wmma_f32_16x16x32_bf16(
              false, a[m2].v, false, b[n2].v, (short)0, acc[m2][n2], false, false);
    }
    __syncthreads();
  }

  // ---- epilogue: gate * (acc + bias) scatter-added into out ----
#pragma unroll
  for (int n2 = 0; n2 < 2; ++n2) {
    const int col = n0 + 32 * wn + 16 * n2 + lidx;
    const float bias = be[e * DOUT + col];
#pragma unroll
    for (int m2 = 0; m2 < 2; ++m2) {
#pragma unroll
      for (int r = 0; r < 8; ++r) {
        const int lr = 32 * wm + 16 * m2 + 8 * sub + r;  // C/D layout: M = r + 8*sub
        const int token = ltok[lr];
        if (token >= 0) {
          const float v = lgate[lr] * (acc[m2][n2][r] + bias);
          unsafeAtomicAdd(out + (size_t)token * DOUT + col, v);  // global_atomic_add_f32
        }
      }
    }
  }
}

// ---------------- launcher --------------------------------------------------
extern "C" void kernel_launch(void* const* d_in, const int* in_sizes, int n_in,
                              void* d_out, int out_size, void* d_ws, size_t ws_size,
                              hipStream_t stream) {
  const float* x  = (const float*)d_in[0];
  const float* Wr = (const float*)d_in[1];
  const float* br = (const float*)d_in[2];
  const float* We = (const float*)d_in[3];
  const float* be = (const float*)d_in[4];

  float* out   = (float*)d_out;
  float* probs = out + (size_t)T_TOK * DOUT;   // tuple: (out, probs) flat

  // carve workspace (~300 KB)
  char* ws = (char*)d_ws;
  auto carve = [&](size_t bytes) {
    char* p = ws;
    ws += (bytes + 255) & ~(size_t)255;
    return p;
  };
  int*   counts    = (int*)  carve(NEXP * sizeof(int));
  int*   cursor    = (int*)  carve(NEXP * sizeof(int));
  int*   off       = (int*)  carve((NEXP + 1) * sizeof(int));
  int*   tok_e     = (int*)  carve((size_t)T_TOK * TOPK * sizeof(int));
  float* tok_p     = (float*)carve((size_t)T_TOK * TOPK * sizeof(float));
  int*   row_token = (int*)  carve((size_t)ROWS_PAD * sizeof(int));
  float* row_gate  = (float*)carve((size_t)ROWS_PAD * sizeof(float));

  const long total_out = (long)T_TOK * DOUT;
  moe_init<<<(int)((total_out + 1023) / 1024), 1024, 0, stream>>>(out, row_token, counts,
                                                                 total_out);
  moe_router<<<T_TOK / 16, 256, 0, stream>>>(x, Wr, br, probs, tok_e, tok_p, counts);
  moe_offsets<<<1, 1, 0, stream>>>(counts, off, cursor);
  moe_place<<<(T_TOK * TOPK) / 256, 256, 0, stream>>>(tok_e, tok_p, off, cursor,
                                                      row_token, row_gate);
  dim3 grid(MAX_MTILES, DOUT / TN);
  moe_expert_gemm<<<grid, 256, 0, stream>>>(x, We, be, row_token, row_gate, off, out);
}
